// LSTM_36618891166279
// MI455X (gfx1250) — compile-verified, matrix operand order checked by reading, not run on hardware
//
#include <hip/hip_runtime.h>

// ---------------------------------------------------------------------------
// LSTM forward + CE loss for MI455X (gfx1250, wave32, WMMA 16x16x32 bf16,
// Tensor Data Mover staging).
//   * Gate weights packed per-WMMA-A-fragment -> one 32B load per lane.
//   * Activations transposed (batch-major, K contiguous); per-K-chunk 2D
//     tile (64 rows x 64B) staged into LDS by TDM (tensor_load_to_lds) with
//     hardware row padding (64B data + 16B pad = bank-friendly 80B rows),
//     double-buffered so the DMA overlaps WMMA.
//   * 64 "gate" blocks own 16 h's x 4 gates; 8 "y" blocks compute the
//     lagged y_{t-1} = Wy @ a_{t-1} so each step is ONE kernel launch.
// ---------------------------------------------------------------------------

typedef __attribute__((ext_vector_type(16))) __bf16 v16bf;
typedef __attribute__((ext_vector_type(8)))  __bf16 v8bf;
typedef __attribute__((ext_vector_type(8)))  float  v8f;
typedef unsigned uint32x4_t __attribute__((ext_vector_type(4)));
typedef unsigned uint32x8_t __attribute__((ext_vector_type(8)));

constexpr int Hc  = 1024;
constexpr int Vc  = 512;
constexpr int Bc  = 64;
constexpr int Tc  = 256;
constexpr int Kg  = Hc + Vc;   // 1536
constexpr int KCG = Kg / 32;   // 48 k-chunks for the gate GEMM
constexpr int KCY = Hc / 32;   // 32 k-chunks for the y GEMM

__device__ __forceinline__ unsigned short f2bf(float x) {
    unsigned u = __float_as_uint(x);
    unsigned r = ((u >> 16) & 1u) + 0x7FFFu;   // round to nearest even
    return (unsigned short)((u + r) >> 16);
}

__device__ __forceinline__ v16bf combine(v8bf lo, v8bf hi) {
    return __builtin_shufflevector(lo, hi, 0, 1, 2, 3, 4, 5, 6, 7,
                                           8, 9, 10, 11, 12, 13, 14, 15);
}

// ---------------------------------------------------------------------------
// TDM: DMA one 2D tile (64 rows x 16 4-byte elems = 64B/row) global -> LDS,
// inserting 16B of LDS padding after every 64B (pad_interval=3 -> 16 DWORDs,
// pad_amount=3 -> 4 DWORDs), matching the [64][40] u16 LDS layout.
// Issued per-wave (EXEC ignored); completion via TENSORcnt on issuing wave.
// ---------------------------------------------------------------------------
__device__ __forceinline__ void tdm_stage_2d(const void* gsrc, unsigned strideU4,
                                             unsigned ldsAddr) {
    unsigned long long ga = (unsigned long long)gsrc;
    uint32x4_t g0 = {
        0x1u,                                            // count=1, user desc
        ldsAddr,                                         // lds_addr (bytes)
        (unsigned)ga,                                    // global_addr[31:0]
        ((unsigned)(ga >> 32) & 0x01FFFFFFu) | 0x80000000u // addr[56:32]|type=2
    };
    uint32x8_t g1 = {
        0x06D20000u,     // data_size=4B | pad_enable | pad_interval=3 | pad_amount=3
        16u << 16,       // tensor_dim0 = 16 elems (64B row)
        64u << 16,       // tensor_dim1 = 64 rows
        16u << 16,       // tile_dim0   = 16 elems
        64u,             // tile_dim1   = 64 rows
        strideU4,        // tensor_dim0_stride (4B units)
        0u, 0u
    };
    asm volatile("tensor_load_to_lds %0, %1" :: "s"(g0), "s"(g1) : "memory");
}

// ---------------------------------------------------------------------------
// Prep kernels (run once per launch; weights then live in the 192MB L2)
// ---------------------------------------------------------------------------

// Pack Wf/Wu/Wcc/Wo into WMMA A-fragment order, gates interleaved per h-block:
// packed row gm: hb = gm/64, gate = (gm/16)%4, hl = gm%16.
__global__ void pack_gate_weights(const float* __restrict__ Wf,
                                  const float* __restrict__ Wu,
                                  const float* __restrict__ Wcc,
                                  const float* __restrict__ Wo,
                                  unsigned short* __restrict__ Wg_pack) {
    int idx = blockIdx.x * blockDim.x + threadIdx.x;
    const int total = 4 * Hc * Kg;
    if (idx >= total) return;
    int e     = idx & 15;
    int lane  = (idx >> 4) & 31;
    int kc    = (idx >> 9) % KCG;
    int mtile = idx / (512 * KCG);
    // A layout: lane<16 holds K {0..7,16..23}, lane>=16 holds K {8..15,24..31}
    int k  = kc * 32 + (e & 7) + ((lane >> 4) << 3) + ((e >> 3) << 4);
    int gm = mtile * 16 + (lane & 15);
    int gate = (gm >> 4) & 3;
    int h    = ((gm >> 6) << 4) | (gm & 15);
    const float* W = (gate == 0) ? Wf : (gate == 1) ? Wu : (gate == 2) ? Wcc : Wo;
    Wg_pack[idx] = f2bf(W[h * Kg + k]);
}

__global__ void pack_y_weights(const float* __restrict__ Wy,
                               unsigned short* __restrict__ Wy_pack) {
    int idx = blockIdx.x * blockDim.x + threadIdx.x;
    const int total = Vc * Hc;
    if (idx >= total) return;
    int e     = idx & 15;
    int lane  = (idx >> 4) & 31;
    int kc    = (idx >> 9) % KCY;
    int mtile = idx / (512 * KCY);
    int k = kc * 32 + (e & 7) + ((lane >> 4) << 3) + ((e >> 3) << 4);
    int m = mtile * 16 + (lane & 15);
    Wy_pack[idx] = f2bf(Wy[m * Hc + k]);
}

__global__ void pack_bias(const float* __restrict__ bf_, const float* __restrict__ bu_,
                          const float* __restrict__ bcc_, const float* __restrict__ bo_,
                          float* __restrict__ bias_pack) {
    int idx = blockIdx.x * blockDim.x + threadIdx.x;
    if (idx >= 4 * Hc) return;
    int hb = idx >> 6, g = (idx >> 4) & 3, hl = idx & 15;
    const float* b = (g == 0) ? bf_ : (g == 1) ? bu_ : (g == 2) ? bcc_ : bo_;
    bias_pack[idx] = b[hb * 16 + hl];
}

// batch_X (B,T,V) f32 -> XtT (T,B,V) bf16 (batch-major, K contiguous)
__global__ void prep_x(const float* __restrict__ X, unsigned short* __restrict__ XtT) {
    int idx = blockIdx.x * blockDim.x + threadIdx.x;
    const int total = Tc * Bc * Vc;
    if (idx >= total) return;
    int v = idx % Vc;
    int b = (idx / Vc) % Bc;
    int t = idx / (Vc * Bc);
    XtT[idx] = f2bf(X[(b * Tc + t) * Vc + v]);
}

// a_prev/c_prev (H,B) -> transposed (B,H); a as bf16, c as f32
__global__ void init_state(const float* __restrict__ a0, const float* __restrict__ c0,
                           unsigned short* __restrict__ aT, float* __restrict__ cT) {
    int idx = blockIdx.x * blockDim.x + threadIdx.x;
    if (idx >= Hc * Bc) return;
    int h = idx / Bc, b = idx % Bc;
    aT[b * Hc + h] = f2bf(a0[idx]);
    cT[b * Hc + h] = c0[idx];
}

// ---------------------------------------------------------------------------
// Core GEMM fragments (wave32 WMMA bf16 -> f32, TDM-staged B operand)
// ---------------------------------------------------------------------------

// y = Wy @ a_{t_y}  (M=512, K=1024, N=64), writes logits rows (b*T + t_y)
__device__ __forceinline__ void y_compute(const unsigned short* __restrict__ Wy_pack,
                                          const float* __restrict__ by,
                                          const unsigned short* __restrict__ aT,
                                          float* __restrict__ logits, int t_y, int yb,
                                          unsigned short (*actS)[64][40]) {
    const int lane  = threadIdx.x & 31;
    const int w     = threadIdx.x >> 5;
    const int m_sub = w & 3;
    const int n0    = (w >> 2) << 1;
    const int mtile = yb * 4 + m_sub;
    const int half  = lane >> 4;
    v8f acc[2] = {};

    if (w == 0) {
        tdm_stage_2d(aT, Hc / 2, (unsigned)(unsigned long long)&actS[0][0][0]);
        __builtin_amdgcn_s_wait_tensorcnt(0);
    }
    __syncthreads();

    for (int kc = 0; kc < KCY; ++kc) {
        if (w == 0 && kc + 1 < KCY) {
            tdm_stage_2d(aT + (kc + 1) * 32, Hc / 2,
                         (unsigned)(unsigned long long)&actS[(kc + 1) & 1][0][0]);
        }
        unsigned short (*cur)[40] = actS[kc & 1];

        const unsigned short* ap = Wy_pack + ((mtile * KCY + kc) * 32 + lane) * 16;
        if (kc + 1 < KCY)
            __builtin_prefetch(Wy_pack + ((mtile * KCY + kc + 1) * 32 + lane) * 16, 0, 0);
        v16bf a = combine(*(const v8bf*)ap, *(const v8bf*)(ap + 8));

        const v8bf* r0 = (const v8bf*)&cur[(n0 + 0) * 16 + (lane & 15)][0];
        const v8bf* r1 = (const v8bf*)&cur[(n0 + 1) * 16 + (lane & 15)][0];
        v16bf b0 = combine(r0[half], r0[half + 2]);
        v16bf b1 = combine(r1[half], r1[half + 2]);

        acc[0] = __builtin_amdgcn_wmma_f32_16x16x32_bf16(false, a, false, b0,
                                                         (short)0, acc[0], false, false);
        acc[1] = __builtin_amdgcn_wmma_f32_16x16x32_bf16(false, a, false, b1,
                                                         (short)0, acc[1], false, false);

        if (w == 0) __builtin_amdgcn_s_wait_tensorcnt(0);
        __syncthreads();
    }

    const int rbase = yb * 64 + m_sub * 16 + (half << 3);
#pragma unroll
    for (int j = 0; j < 2; ++j) {
        int bcol = (n0 + j) * 16 + (lane & 15);
#pragma unroll
        for (int vv = 0; vv < 8; ++vv) {
            int row = rbase + vv;
            logits[((size_t)bcol * Tc + t_y) * Vc + row] = acc[j][vv] + by[row];
        }
    }
}

// z = Wg @ [a_{t-1}; x_t] for this block's 16 h's x 4 gates, then gate math.
__device__ __forceinline__ void gate_compute(const unsigned short* __restrict__ Wg_pack,
                                             const float* __restrict__ bias_pack,
                                             const unsigned short* __restrict__ XtT,
                                             const unsigned short* __restrict__ aT_prev,
                                             unsigned short* __restrict__ aT_next,
                                             float* __restrict__ cT, int t, int hb,
                                             unsigned short (*actS)[64][40],
                                             float (*zbuf)[64]) {
    const int lane  = threadIdx.x & 31;
    const int w     = threadIdx.x >> 5;
    const int m_sub = w & 3;              // == gate index for this block
    const int n0    = (w >> 2) << 1;
    const int mtile = hb * 4 + m_sub;
    const int half  = lane >> 4;
    v8f acc[2] = {};
    const unsigned short* xbase = XtT + (size_t)t * Bc * Vc;

    if (w == 0) {
        tdm_stage_2d(aT_prev, Hc / 2, (unsigned)(unsigned long long)&actS[0][0][0]);
        __builtin_amdgcn_s_wait_tensorcnt(0);
    }
    __syncthreads();

    for (int kc = 0; kc < KCG; ++kc) {
        if (w == 0 && kc + 1 < KCG) {
            const int k1 = (kc + 1) * 32;
            const unsigned short* nsrc =
                (k1 < Hc) ? (aT_prev + k1) : (xbase + (k1 - Hc));
            unsigned nstr = (k1 < Hc) ? (unsigned)(Hc / 2) : (unsigned)(Vc / 2);
            tdm_stage_2d(nsrc, nstr,
                         (unsigned)(unsigned long long)&actS[(kc + 1) & 1][0][0]);
        }
        unsigned short (*cur)[40] = actS[kc & 1];

        const unsigned short* ap = Wg_pack + ((mtile * KCG + kc) * 32 + lane) * 16;
        if (kc + 1 < KCG)
            __builtin_prefetch(Wg_pack + ((mtile * KCG + kc + 1) * 32 + lane) * 16, 0, 0);
        v16bf a = combine(*(const v8bf*)ap, *(const v8bf*)(ap + 8));

        const v8bf* r0 = (const v8bf*)&cur[(n0 + 0) * 16 + (lane & 15)][0];
        const v8bf* r1 = (const v8bf*)&cur[(n0 + 1) * 16 + (lane & 15)][0];
        v16bf b0 = combine(r0[half], r0[half + 2]);
        v16bf b1 = combine(r1[half], r1[half + 2]);

        acc[0] = __builtin_amdgcn_wmma_f32_16x16x32_bf16(false, a, false, b0,
                                                         (short)0, acc[0], false, false);
        acc[1] = __builtin_amdgcn_wmma_f32_16x16x32_bf16(false, a, false, b1,
                                                         (short)0, acc[1], false, false);

        if (w == 0) __builtin_amdgcn_s_wait_tensorcnt(0);
        __syncthreads();
    }

    // Spill z tile (gate-major rows) so all 4 gates of each h are visible.
    const int zrow0 = m_sub * 16 + (half << 3);
#pragma unroll
    for (int j = 0; j < 2; ++j) {
        int col = (n0 + j) * 16 + (lane & 15);
#pragma unroll
        for (int vv = 0; vv < 8; ++vv)
            zbuf[zrow0 + vv][col] = acc[j][vv];
    }
    __syncthreads();

    // Elementwise gates: each thread owns (batch n, 4 consecutive h's).
    const int n  = threadIdx.x & 63;
    const int hq = threadIdx.x >> 6;
    float cvals[4];
    unsigned short avals[4];
#pragma unroll
    for (int i = 0; i < 4; ++i) {
        int hl = hq * 4 + i;
        float zf = zbuf[hl][n]      + bias_pack[hb * 64 + hl];
        float zu = zbuf[16 + hl][n] + bias_pack[hb * 64 + 16 + hl];
        float zc = zbuf[32 + hl][n] + bias_pack[hb * 64 + 32 + hl];
        float zo = zbuf[48 + hl][n] + bias_pack[hb * 64 + 48 + hl];
        float fg = 1.f / (1.f + __expf(-zf));
        float ug = 1.f / (1.f + __expf(-zu));
        float cc = tanhf(zc);
        float og = 1.f / (1.f + __expf(-zo));
        int hg   = hb * 16 + hl;
        float c  = fg * cT[n * Hc + hg] + ug * cc;
        cvals[i] = c;
        avals[i] = f2bf(og * tanhf(c));
    }
    const int hgBase = hb * 16 + hq * 4;
    *(float4*)&cT[n * Hc + hgBase] = make_float4(cvals[0], cvals[1], cvals[2], cvals[3]);
    *(ushort4*)&aT_next[n * Hc + hgBase] =
        make_ushort4(avals[0], avals[1], avals[2], avals[3]);
}

// ---------------------------------------------------------------------------
// Per-step kernel: blocks 0..63 gates (t), blocks 64..71 logits (t-1)
// ---------------------------------------------------------------------------
__global__ void __launch_bounds__(256) lstm_step(const unsigned short* Wg_pack,
                                                 const float* bias_pack,
                                                 const unsigned short* Wy_pack,
                                                 const float* by,
                                                 const unsigned short* XtT,
                                                 const unsigned short* aT_prev,
                                                 unsigned short* aT_next,
                                                 float* cT, float* logits, int t) {
    __shared__ __align__(16) unsigned short actS[2][64][40]; // double-buffered stage
    __shared__ float zbuf[64][64];
    if (blockIdx.x < 64) {
        gate_compute(Wg_pack, bias_pack, XtT, aT_prev, aT_next, cT, t,
                     blockIdx.x, actS, zbuf);
    } else {
        if (t == 0) return;
        y_compute(Wy_pack, by, aT_prev, logits, t - 1, blockIdx.x - 64, actS);
    }
}

__global__ void __launch_bounds__(256) y_final(const unsigned short* Wy_pack,
                                               const float* by,
                                               const unsigned short* aT,
                                               float* logits) {
    __shared__ __align__(16) unsigned short actS[2][64][40];
    y_compute(Wy_pack, by, aT, logits, Tc - 1, blockIdx.x, actS);
}

// ---------------------------------------------------------------------------
// Deterministic cross-entropy (no float atomics -> graph-replay stable)
// ---------------------------------------------------------------------------
__global__ void row_loss_kernel(const float* __restrict__ logits,
                                const int* __restrict__ labels,
                                float* __restrict__ row_loss) {
    const int r = blockIdx.x;
    const float* row = logits + (size_t)r * Vc;
    const int tid = threadIdx.x;
    __shared__ float red[256];
    float l0 = row[tid], l1 = row[tid + 256];
    red[tid] = fmaxf(l0, l1);
    __syncthreads();
    for (int s = 128; s > 0; s >>= 1) {
        if (tid < s) red[tid] = fmaxf(red[tid], red[tid + s]);
        __syncthreads();
    }
    float m = red[0];
    __syncthreads();
    red[tid] = __expf(l0 - m) + __expf(l1 - m);
    __syncthreads();
    for (int s = 128; s > 0; s >>= 1) {
        if (tid < s) red[tid] += red[tid + s];
        __syncthreads();
    }
    if (tid == 0) {
        int lbl = labels[r];
        row_loss[r] = -(row[lbl] - m - __logf(red[0]));
    }
}

__global__ void loss_reduce(const float* __restrict__ row_loss, float* out_loss) {
    __shared__ float red[256];
    const int tid = threadIdx.x;
    float s = 0.f;
    for (int i = tid; i < Bc * Tc; i += 256) s += row_loss[i];
    red[tid] = s;
    __syncthreads();
    for (int st = 128; st > 0; st >>= 1) {
        if (tid < st) red[tid] += red[tid + st];
        __syncthreads();
    }
    if (tid == 0) *out_loss = red[0] * (1.f / (float)(Bc * Tc));
}

// ---------------------------------------------------------------------------
extern "C" void kernel_launch(void* const* d_in, const int* in_sizes, int n_in,
                              void* d_out, int out_size, void* d_ws, size_t ws_size,
                              hipStream_t stream) {
    (void)in_sizes; (void)n_in; (void)out_size; (void)ws_size;
    const float* X    = (const float*)d_in[0];
    const int*   Y    = (const int*)d_in[1];
    const float* Wf   = (const float*)d_in[2];
    const float* bf_  = (const float*)d_in[3];
    const float* Wu   = (const float*)d_in[4];
    const float* bu_  = (const float*)d_in[5];
    const float* Wcc  = (const float*)d_in[6];
    const float* bcc_ = (const float*)d_in[7];
    const float* Wo   = (const float*)d_in[8];
    const float* bo_  = (const float*)d_in[9];
    const float* Wy   = (const float*)d_in[10];
    const float* by   = (const float*)d_in[11];
    const float* a0   = (const float*)d_in[12];
    const float* c0   = (const float*)d_in[13];

    float* logits = (float*)d_out;
    float* loss   = logits + (size_t)Bc * Tc * Vc;

    char* ws = (char*)d_ws;
    auto alloc = [&](size_t bytes) -> char* {
        char* p = ws;
        ws += (bytes + 255) & ~(size_t)255;
        return p;
    };
    unsigned short* Wg_pack = (unsigned short*)alloc((size_t)4 * Hc * Kg * 2);
    unsigned short* Wy_pack = (unsigned short*)alloc((size_t)Vc * Hc * 2);
    float*          bias_pk = (float*)alloc((size_t)4 * Hc * 4);
    unsigned short* XtT     = (unsigned short*)alloc((size_t)Tc * Bc * Vc * 2);
    unsigned short* aA      = (unsigned short*)alloc((size_t)Bc * Hc * 2);
    unsigned short* aB      = (unsigned short*)alloc((size_t)Bc * Hc * 2);
    float*          cT      = (float*)alloc((size_t)Bc * Hc * 4);
    float*          rloss   = (float*)alloc((size_t)Bc * Tc * 4);

    pack_gate_weights<<<(4 * Hc * Kg + 255) / 256, 256, 0, stream>>>(Wf, Wu, Wcc, Wo, Wg_pack);
    pack_y_weights<<<(Vc * Hc + 255) / 256, 256, 0, stream>>>(Wy, Wy_pack);
    pack_bias<<<(4 * Hc + 255) / 256, 256, 0, stream>>>(bf_, bu_, bcc_, bo_, bias_pk);
    prep_x<<<(Tc * Bc * Vc + 255) / 256, 256, 0, stream>>>(X, XtT);
    init_state<<<(Hc * Bc + 255) / 256, 256, 0, stream>>>(a0, c0, aA, cT);

    unsigned short* bufs[2] = {aA, aB};
    for (int t = 0; t < Tc; ++t) {
        lstm_step<<<72, 256, 0, stream>>>(Wg_pack, bias_pk, Wy_pack, by, XtT,
                                          bufs[t & 1], bufs[(t + 1) & 1], cT, logits, t);
    }
    y_final<<<8, 256, 0, stream>>>(Wy_pack, by, bufs[Tc & 1], logits);

    row_loss_kernel<<<Bc * Tc, 256, 0, stream>>>(logits, Y, rloss);
    loss_reduce<<<1, 256, 0, stream>>>(rloss, loss);
}
